// BpfoldModule_14740327759951
// MI455X (gfx1250) — compile-verified, matrix operand order checked by reading, not run on hardware
//
#include <hip/hip_runtime.h>
#include <hip/hip_bf16.h>
#include <math.h>
#include <stdint.h>

// ---------------------------------------------------------------------------
// BpfoldModule transformer layer for MI455X (gfx1250), bf16 WMMA pipeline.
// B=2, S=1024, D=1024, H=16, HD=64, I=4096. fp32 in/out, bf16 matrix cores.
// All matrix operands are arranged contiguous-in-K so every WMMA fragment is
// two 16-byte loads (global_load_b128 / ds_load_b128) + a register shuffle.
// ---------------------------------------------------------------------------

typedef __bf16 bf16_t;
typedef __attribute__((ext_vector_type(16))) __bf16 v16bf;
typedef __attribute__((ext_vector_type(8)))  __bf16 v8bf;
typedef __attribute__((ext_vector_type(8)))  float  v8f;
typedef __attribute__((ext_vector_type(4)))  unsigned int u32x4;
typedef __attribute__((ext_vector_type(8)))  int i32x8;
typedef __attribute__((ext_vector_type(4)))  int i32x4;

#define B_  2
#define S_  1024
#define D_  1024
#define H_  16
#define HD_ 64
#define I_  4096

#if defined(__has_builtin)
#if __has_builtin(__builtin_amdgcn_tensor_load_to_lds) && \
    __has_builtin(__builtin_amdgcn_s_wait_tensorcnt)
#define HAVE_TDM 1
#endif
#endif

__device__ __forceinline__ v8f wmma_bf16(v16bf a, v16bf b, v8f c) {
  return __builtin_amdgcn_wmma_f32_16x16x32_bf16(
      /*neg_a=*/false, a, /*neg_b=*/false, b,
      /*c_mod=*/(short)0, c, /*reuse_a=*/false, /*reuse_b=*/false);
}

// 16-bit A/B fragment (ISA 7.12.2): lane's 16 elements are K = 8*lh + [0,8)
// and K = 16 + 8*lh + [0,8).  Fetch as two b128 loads from a contiguous-in-K
// row and merge in registers.
__device__ __forceinline__ v16bf ld_frag(const bf16_t* row_base, int lh) {
  v8bf lo = *(const v8bf*)(row_base + (lh << 3));
  v8bf hi = *(const v8bf*)(row_base + 16 + (lh << 3));
  return __builtin_shufflevector(lo, hi, 0, 1, 2, 3, 4, 5, 6, 7,
                                 8, 9, 10, 11, 12, 13, 14, 15);
}

// ---------------------------------------------------------------------------
// fp32 [R,C] -> bf16 transpose [C,R] (weights become B^T, contiguous in K)
// ---------------------------------------------------------------------------
__global__ __launch_bounds__(256)
void transpose_cvt(const float* __restrict__ in, bf16_t* __restrict__ out,
                   int R, int C) {
  __shared__ float tile[32][33];
  const int c0 = blockIdx.x * 32;
  const int r0 = blockIdx.y * 32;
  for (int i = threadIdx.y; i < 32; i += 8)
    tile[i][threadIdx.x] = in[(size_t)(r0 + i) * C + c0 + threadIdx.x];
  __syncthreads();
  for (int i = threadIdx.y; i < 32; i += 8)
    out[(size_t)(c0 + i) * R + r0 + threadIdx.x] = (bf16_t)tile[threadIdx.x][i];
}

// ---------------------------------------------------------------------------
// LayerNorm over D=1024, one row per block, bf16 output
// ---------------------------------------------------------------------------
__global__ __launch_bounds__(256)
void ln_kernel(const float* __restrict__ x, const float* __restrict__ g,
               const float* __restrict__ bb, bf16_t* __restrict__ out) {
  __shared__ float red[256];
  __shared__ float red2[256];
  const size_t row = blockIdx.x;
  const float* xr = x + row * D_;
  float s = 0.f, s2 = 0.f;
  for (int c = threadIdx.x; c < D_; c += 256) {
    float v = xr[c];
    s += v;
    s2 += v * v;
  }
  red[threadIdx.x] = s;
  red2[threadIdx.x] = s2;
  __syncthreads();
  for (int off = 128; off > 0; off >>= 1) {
    if ((int)threadIdx.x < off) {
      red[threadIdx.x]  += red[threadIdx.x + off];
      red2[threadIdx.x] += red2[threadIdx.x + off];
    }
    __syncthreads();
  }
  const float mean = red[0] * (1.0f / D_);
  const float var  = red2[0] * (1.0f / D_) - mean * mean;
  const float rstd = rsqrtf(var + 1e-5f);
  for (int c = threadIdx.x; c < D_; c += 256)
    out[row * D_ + c] = (bf16_t)((xr[c] - mean) * rstd * g[c] + bb[c]);
}

// ---------------------------------------------------------------------------
// LDS-free bf16 GEMM: C[M,N] = A[M,K] x Bt[N,K]^T (+epilogue).
// Block = 8 waves; each wave owns a 32x64 strip: 2 m-subtiles x 4 n-subtiles
// = 8 f32 WMMA accumulators; per K-chunk: 12 b128 loads, 8 WMMA.
// Weights/activations are L2-resident so global re-reads are cheap.
// MODE 0: +qkv_b, scatter q/k (row-major [S,hd]) and v transposed [hd,S]
// MODE 1: +b_in, exact GELU, bf16 out
// MODE 2: +b_out, +residual, f32 out
// ---------------------------------------------------------------------------
template <int MODE>
__global__ __launch_bounds__(256)
void gemm_kernel(const bf16_t* __restrict__ A, const bf16_t* __restrict__ Bt,
                 int M, int N, int K,
                 const float* __restrict__ bias,
                 float* __restrict__ fout, const float* __restrict__ resid,
                 bf16_t* __restrict__ bout,
                 bf16_t* __restrict__ qout, bf16_t* __restrict__ kout,
                 bf16_t* __restrict__ vout) {
  const int lane = threadIdx.x & 31;
  const int wave = threadIdx.x >> 5;
  const int ln15 = lane & 15;
  const int lh = lane >> 4;
  const int mBase = blockIdx.x * 256 + wave * 32;  // 32 rows per wave
  const int nBase = blockIdx.y * 64;               // 64 cols per wave

  const bf16_t* a0 = A + (size_t)(mBase + ln15) * K;        // m-subtile 0 row
  const bf16_t* a1 = A + (size_t)(mBase + 16 + ln15) * K;   // m-subtile 1 row
  const bf16_t* b0 = Bt + (size_t)(nBase + ln15) * K;
  const bf16_t* b1 = Bt + (size_t)(nBase + 16 + ln15) * K;
  const bf16_t* b2 = Bt + (size_t)(nBase + 32 + ln15) * K;
  const bf16_t* b3 = Bt + (size_t)(nBase + 48 + ln15) * K;

  v8f acc[2][4] = {};
  for (int kc = 0; kc < K; kc += 32) {
    const v16bf af0 = ld_frag(a0 + kc, lh);
    const v16bf af1 = ld_frag(a1 + kc, lh);
    const v16bf bf0 = ld_frag(b0 + kc, lh);
    acc[0][0] = wmma_bf16(af0, bf0, acc[0][0]);
    acc[1][0] = wmma_bf16(af1, bf0, acc[1][0]);
    const v16bf bf1 = ld_frag(b1 + kc, lh);
    acc[0][1] = wmma_bf16(af0, bf1, acc[0][1]);
    acc[1][1] = wmma_bf16(af1, bf1, acc[1][1]);
    const v16bf bf2 = ld_frag(b2 + kc, lh);
    acc[0][2] = wmma_bf16(af0, bf2, acc[0][2]);
    acc[1][2] = wmma_bf16(af1, bf2, acc[1][2]);
    const v16bf bf3 = ld_frag(b3 + kc, lh);
    acc[0][3] = wmma_bf16(af0, bf3, acc[0][3]);
    acc[1][3] = wmma_bf16(af1, bf3, acc[1][3]);
  }

  // epilogue (C layout: lane -> N = lane&15, reg r -> M = r + 8*(lane>>4))
#pragma unroll
  for (int ms = 0; ms < 2; ++ms) {
#pragma unroll
    for (int ns = 0; ns < 4; ++ns) {
      const int n_g = nBase + (ns << 4) + ln15;
#pragma unroll
      for (int r = 0; r < 8; ++r) {
        const int m_g = mBase + (ms << 4) + r + (lh << 3);
        float v = acc[ms][ns][r] + bias[n_g];
        if (MODE == 0) {
          const int head = n_g / 192;     // 3*HD per head
          const int within = n_g % 192;
          const int part = within >> 6;   // 0=q 1=k 2=v
          const int hd = within & 63;
          const int bidx = m_g / S_;
          const int sidx = m_g & (S_ - 1);
          if (part == 2)                  // v stored transposed: [B,H,hd,S]
            vout[(size_t)((bidx * H_ + head) * HD_ + hd) * S_ + sidx] = (bf16_t)v;
          else {
            bf16_t* dst = (part == 0) ? qout : kout;
            dst[(size_t)((bidx * H_ + head) * S_ + sidx) * HD_ + hd] = (bf16_t)v;
          }
        } else if (MODE == 1) {
          const float gl = 0.5f * v * (1.0f + erff(v * 0.70710678118f));
          bout[(size_t)m_g * N + n_g] = (bf16_t)gl;
        } else {
          fout[(size_t)m_g * N + n_g] = v + resid[(size_t)m_g * N + n_g];
        }
      }
    }
  }
}

// ---------------------------------------------------------------------------
// Attention: one wave per (b, h, 16-row q-tile).
//   Q tile staged to LDS by the Tensor Data Mover (TDM) + s_wait_tensorcnt.
//   S1 = (Q Kt)/8 + alibi + bias, masked  -> 16x1024 f32 scores in LDS
//   softmax (wave shfl reductions), probs re-packed bf16 in-place
//   S2 = P V (V transposed [hd,S] => contiguous-in-key B frags)
//   out + out_bias + residual(x) -> attn_out f32
// ---------------------------------------------------------------------------
__global__ __launch_bounds__(32)
void attn_kernel(const bf16_t* __restrict__ q, const bf16_t* __restrict__ k,
                 const bf16_t* __restrict__ vT,
                 const float* __restrict__ attn_bias,
                 const unsigned char* __restrict__ mask,
                 const float* __restrict__ out_bias,
                 const float* __restrict__ x,
                 float* __restrict__ attn_out) {
  __shared__ __align__(16) float sc[16 * 1024];   // 64 KB scores
  __shared__ __align__(16) bf16_t qs[16 * 64];    // 2 KB Q tile (TDM dest)
  bf16_t* pr = reinterpret_cast<bf16_t*>(sc);     // bf16 probs alias

  const int lane = threadIdx.x;
  const int ln15 = lane & 15;
  const int lh = lane >> 4;
  const int wg = blockIdx.x;            // b * H * (S/16)
  const int qt = wg & 63;               // q-tile index (S/16 = 64)
  const int h = (wg >> 6) & (H_ - 1);
  const int b = wg >> 10;
  const size_t headoff = (size_t)(b * H_ + h) * S_;
  // ALiBi: 8 real heads with slope 2^-(i+1), padded heads slope 0
  const float slope = (h < 8) ? exp2f(-(float)(h + 1)) : 0.0f;

  // ---- stage Q tile (16x64 bf16 = 1024 contiguous elements) via TDM ----
#if defined(HAVE_TDM)
  {
    const unsigned long long ga =
        (unsigned long long)(uintptr_t)(q + (headoff + qt * 16) * HD_);
    const unsigned int la = (unsigned int)(uintptr_t)&qs[0];  // LDS offset
    u32x4 g0;
    g0[0] = 1u;                                   // count=1 (valid user D#)
    g0[1] = la;                                   // lds_addr
    g0[2] = (unsigned int)ga;                     // global_addr[31:0]
    g0[3] = (unsigned int)((ga >> 32) & 0x1FFFFFFull) | (2u << 30);  // type=2
    i32x8 g1;
    g1[0] = 1 << 16;          // data_size=1 (2 bytes)
    g1[1] = 2048 << 16;       // tensor_dim0[15:0]=2048
    g1[2] = 1 << 16;          // tensor_dim1=1
    g1[3] = 2048 << 16;       // tile_dim0=2048 (1-D tile)
    g1[4] = 1;                // tile_dim1=1
    g1[5] = 2048;             // tensor_dim0_stride[31:0]
    g1[6] = 0;
    g1[7] = 0;
    const i32x4 gz4 = {0, 0, 0, 0};               // groups 2/3 unused (2-D)
    const i32x8 gz8 = {0, 0, 0, 0, 0, 0, 0, 0};
    __builtin_amdgcn_tensor_load_to_lds(g0, g1, gz4, gz4, gz8, 0);
    __builtin_amdgcn_s_wait_tensorcnt(0);
  }
#else
  {
    const v8bf* src = (const v8bf*)(q + (headoff + qt * 16) * HD_);
    v8bf* dst = (v8bf*)qs;
    for (int i = lane; i < (16 * 64) / 8; i += 32) dst[i] = src[i];
  }
#endif

  // ---- Q fragments from LDS (hd = 0..31, 32..63) ----
  const v16bf aq0 = ld_frag(qs + ln15 * 64, lh);
  const v16bf aq1 = ld_frag(qs + ln15 * 64 + 32, lh);

  // ---- scores: 64 key tiles of 16 ----
  for (int j = 0; j < 64; ++j) {
    const bf16_t* krow = k + (headoff + j * 16 + ln15) * HD_;
    const v16bf bk0 = ld_frag(krow, lh);       // B[k=hd][n=key] == k[key][hd]
    const v16bf bk1 = ld_frag(krow + 32, lh);
    v8f acc = {};
    acc = wmma_bf16(aq0, bk0, acc);
    acc = wmma_bf16(aq1, bk1, acc);

    const int keycol = j * 16 + ln15;
    const bool km = mask[b * S_ + keycol] != 0;
#pragma unroll
    for (int r = 0; r < 8; ++r) {
      const int m = r + (lh << 3);
      const int qrow_g = qt * 16 + m;
      float val = acc[r] * 0.125f                         // 1/sqrt(hd), T=1
                  - slope * fabsf((float)(keycol - qrow_g))
                  + attn_bias[(headoff + qrow_g) * S_ + keycol];
      if (!km) val = -3.402823466e38f;
      sc[m * 1024 + keycol] = val;
    }
  }

  // ---- softmax per row, wave-wide ----
  for (int m = 0; m < 16; ++m) {
    const int qrow_g = qt * 16 + m;
    float mx = -3.402823466e38f;
    for (int c = lane; c < 1024; c += 32) mx = fmaxf(mx, sc[m * 1024 + c]);
#pragma unroll
    for (int off = 16; off > 0; off >>= 1) mx = fmaxf(mx, __shfl_xor(mx, off, 32));
    float sum = 0.0f;
    for (int c = lane; c < 1024; c += 32) {
      const float e = __expf(sc[m * 1024 + c] - mx);
      sc[m * 1024 + c] = e;
      sum += e;
    }
#pragma unroll
    for (int off = 16; off > 0; off >>= 1) sum += __shfl_xor(sum, off, 32);
    const bool qm = mask[b * S_ + qrow_g] != 0;
    const float inv = qm ? (1.0f / sum) : 0.0f;  // att *= query mask
    // bf16 writes for row m land strictly below f32 reads of rows >= m
    for (int c = lane; c < 1024; c += 32)
      pr[m * 1024 + c] = (bf16_t)(sc[m * 1024 + c] * inv);
  }

  // ---- out = P x V : 32 K-chunks x 4 hd sub-tiles, all b128 frag loads ----
  v8f oacc[4] = {};
  const bf16_t* prow = pr + ln15 * 1024;  // P A-frag row (LDS)
  const bf16_t* vh = vT + (size_t)(b * H_ + h) * HD_ * S_;
  for (int kc = 0; kc < 32; ++kc) {
    const v16bf ap = ld_frag(prow + kc * 32, lh);
#pragma unroll
    for (int ns = 0; ns < 4; ++ns) {
      // B[k=key][n=hd] == vT[hd][key] -> contiguous in key
      const v16bf bv = ld_frag(vh + (size_t)((ns << 4) + ln15) * S_ + kc * 32, lh);
      oacc[ns] = wmma_bf16(ap, bv, oacc[ns]);
    }
  }

  // ---- epilogue: + out_bias + residual(x) ----
#pragma unroll
  for (int ns = 0; ns < 4; ++ns) {
    const int ncol = h * 64 + (ns << 4) + ln15;
#pragma unroll
    for (int r = 0; r < 8; ++r) {
      const int m = r + (lh << 3);
      const size_t row = (size_t)(b * S_ + qt * 16 + m);
      attn_out[row * D_ + ncol] =
          oacc[ns][r] + out_bias[ncol] + x[row * D_ + ncol];
    }
  }
}

// ---------------------------------------------------------------------------
// Host-side orchestration
// ---------------------------------------------------------------------------
extern "C" void kernel_launch(void* const* d_in, const int* in_sizes, int n_in,
                              void* d_out, int out_size, void* d_ws,
                              size_t ws_size, hipStream_t stream) {
  const float* x        = (const float*)d_in[0];
  const float* abias    = (const float*)d_in[1];
  const unsigned char* amask = (const unsigned char*)d_in[2];
  const float* ln1_g    = (const float*)d_in[3];
  const float* ln1_b    = (const float*)d_in[4];
  const float* qkv_w    = (const float*)d_in[5];
  const float* qkv_b    = (const float*)d_in[6];
  const float* out_bias = (const float*)d_in[7];
  const float* ln2_g    = (const float*)d_in[8];
  const float* ln2_b    = (const float*)d_in[9];
  const float* w_in     = (const float*)d_in[10];
  const float* b_in     = (const float*)d_in[11];
  const float* w_out    = (const float*)d_in[12];
  const float* b_out    = (const float*)d_in[13];
  float* out = (float*)d_out;

  char* ws = (char*)d_ws;
  size_t off = 0;
  auto alloc = [&](size_t bytes) -> void* {
    void* p = ws + off;
    off = (off + bytes + 255) & ~(size_t)255;
    return p;
  };

  const int M = B_ * S_;  // 2048
  bf16_t* qkv_wT = (bf16_t*)alloc((size_t)D_ * 3 * D_ * 2);  // [3D, D]
  bf16_t* w_inT  = (bf16_t*)alloc((size_t)D_ * I_ * 2);      // [I, D]
  bf16_t* w_outT = (bf16_t*)alloc((size_t)I_ * D_ * 2);      // [D, I]
  bf16_t* h1     = (bf16_t*)alloc((size_t)M * D_ * 2);
  bf16_t* qb     = (bf16_t*)alloc((size_t)B_ * H_ * S_ * HD_ * 2);
  bf16_t* kb     = (bf16_t*)alloc((size_t)B_ * H_ * S_ * HD_ * 2);
  bf16_t* vTb    = (bf16_t*)alloc((size_t)B_ * H_ * S_ * HD_ * 2);  // [B,H,hd,S]
  float*  attn_o = (float*)alloc((size_t)M * D_ * 4);
  bf16_t* h2     = (bf16_t*)alloc((size_t)M * D_ * 2);
  bf16_t* ff     = (bf16_t*)alloc((size_t)M * I_ * 2);
  (void)ws_size; (void)in_sizes; (void)n_in; (void)out_size;

  // weight transpose + bf16 conversion (B^T layout, contiguous in K)
  {
    dim3 blk(32, 8);
    dim3 g1((3 * D_) / 32, D_ / 32);
    transpose_cvt<<<g1, blk, 0, stream>>>(qkv_w, qkv_wT, D_, 3 * D_);
    dim3 g2(I_ / 32, D_ / 32);
    transpose_cvt<<<g2, blk, 0, stream>>>(w_in, w_inT, D_, I_);
    dim3 g3(D_ / 32, I_ / 32);
    transpose_cvt<<<g3, blk, 0, stream>>>(w_out, w_outT, I_, D_);
  }

  // LN1 -> h1 (bf16)
  ln_kernel<<<M, 256, 0, stream>>>(x, ln1_g, ln1_b, h1);

  // QKV projection: [2048,1024] x [1024,3072] -> q/k row-major, v transposed
  {
    dim3 g(M / 256, (3 * D_) / 64);
    gemm_kernel<0><<<g, 256, 0, stream>>>(h1, qkv_wT, M, 3 * D_, D_, qkv_b,
                                          nullptr, nullptr, nullptr,
                                          qb, kb, vTb);
  }

  // fused attention (TDM Q-stage, scores+alibi+bias+mask, softmax, PV, resid)
  attn_kernel<<<B_ * H_ * (S_ / 16), 32, 0, stream>>>(qb, kb, vTb, abias,
                                                      amask, out_bias, x,
                                                      attn_o);

  // LN2 -> h2 (bf16)
  ln_kernel<<<M, 256, 0, stream>>>(attn_o, ln2_g, ln2_b, h2);

  // FFN1 + GELU: [2048,1024] x [1024,4096] -> ff (bf16)
  {
    dim3 g(M / 256, I_ / 64);
    gemm_kernel<1><<<g, 256, 0, stream>>>(h2, w_inT, M, I_, D_, b_in,
                                          nullptr, nullptr, ff, nullptr,
                                          nullptr, nullptr);
  }

  // FFN2 + bias + residual: [2048,4096] x [4096,1024] -> out (f32)
  {
    dim3 g(M / 256, D_ / 64);
    gemm_kernel<2><<<g, 256, 0, stream>>>(ff, w_outT, M, D_, I_, b_out,
                                          out, attn_o, nullptr, nullptr,
                                          nullptr, nullptr);
  }
}
